// Int8DynActInt8WeightLinear_5952824672396
// MI455X (gfx1250) — 8-bit matrix operand order repaired and checked by reading, compiled, not run on hardware
//
#include <hip/hip_runtime.h>
#include <stdint.h>

typedef __attribute__((ext_vector_type(8))) int   v8i;
typedef __attribute__((ext_vector_type(4))) int   v4i;
typedef __attribute__((ext_vector_type(2))) int   v2i;
typedef __attribute__((ext_vector_type(4))) float v4f;

#define TOKENS 8192
#define IN_F   4096
#define OUT_F  11008
#define QMINF  (-128.0f)
#define QMAXF  (127.0f)
#define EPSF   1.1920928955078125e-07f

__global__ __launch_bounds__(256) void quant_kernel(
    const float* __restrict__ x, int8_t* __restrict__ q,
    float* __restrict__ sx, float* __restrict__ zp, int* __restrict__ sumq)
{
    __shared__ float redf[256];
    __shared__ int   redi[256];
    const int t   = threadIdx.x;
    const int row = blockIdx.x;
    const float* xr = x + (size_t)row * IN_F;

    v4f v[4];
    const v4f* xv = (const v4f*)xr + t * 4;
    float mn = 3.402823466e+38f, mx = -3.402823466e+38f;
#pragma unroll
    for (int i = 0; i < 4; ++i) {
        v[i] = xv[i];
        mn = fminf(mn, fminf(fminf(v[i].x, v[i].y), fminf(v[i].z, v[i].w)));
        mx = fmaxf(mx, fmaxf(fmaxf(v[i].x, v[i].y), fmaxf(v[i].z, v[i].w)));
    }
    redf[t] = mn; __syncthreads();
    for (int s = 128; s > 0; s >>= 1) { if (t < s) redf[t] = fminf(redf[t], redf[t + s]); __syncthreads(); }
    const float minv = redf[0]; __syncthreads();
    redf[t] = mx; __syncthreads();
    for (int s = 128; s > 0; s >>= 1) { if (t < s) redf[t] = fmaxf(redf[t], redf[t + s]); __syncthreads(); }
    const float maxv = redf[0]; __syncthreads();

    const float min_neg = fminf(minv, 0.0f);
    const float max_pos = fmaxf(maxv, 0.0f);
    float scale = (max_pos - min_neg) * (1.0f / 255.0f);
    scale = fmaxf(scale, EPSF);
    const float desc_min = min_neg / scale;
    const float desc_max = max_pos / scale;
    float zpf = ((QMINF + desc_min) + (QMAXF + desc_max) > 0.0f) ? (QMINF - desc_min)
                                                                 : (QMAXF - desc_max);
    zpf = fminf(fmaxf(rintf(zpf), QMINF), QMAXF);

    union { int8_t b[16]; v4i v; } u;
    int local_sum = 0;
#pragma unroll
    for (int i = 0; i < 4; ++i) {
        const float* e = (const float*)&v[i];
#pragma unroll
        for (int j = 0; j < 4; ++j) {
            float qf = fminf(fmaxf(rintf(e[j] / scale) + zpf, QMINF), QMAXF);
            int   qi = (int)qf;
            u.b[i * 4 + j] = (int8_t)qi;
            local_sum += qi;
        }
    }
    *(v4i*)(q + (size_t)row * IN_F + t * 16) = u.v;

    redi[t] = local_sum; __syncthreads();
    for (int s = 128; s > 0; s >>= 1) { if (t < s) redi[t] += redi[t + s]; __syncthreads(); }
    if (t == 0) { sx[row] = scale; zp[row] = zpf; sumq[row] = redi[0]; }
}

__global__ __launch_bounds__(256) void rowsum_kernel(
    const int8_t* __restrict__ w, int* __restrict__ rsw)
{
    __shared__ int redi[256];
    const int t   = threadIdx.x;
    const int row = blockIdx.x;
    const v4i wv = *((const v4i*)(w + (size_t)row * IN_F) + t);
    int s = 0;
#pragma unroll
    for (int i = 0; i < 4; ++i) {
        const int d = wv[i];
        s += (int)(int8_t)(d) + (int)(int8_t)(d >> 8) +
             (int)(int8_t)(d >> 16) + (int)(int8_t)(d >> 24);
    }
    redi[t] = s; __syncthreads();
    for (int k = 128; k > 0; k >>= 1) { if (t < k) redi[t] += redi[t + k]; __syncthreads(); }
    if (t == 0) rsw[row] = redi[0];
}

__global__ __launch_bounds__(256) void gemm_kernel(
    const int8_t* __restrict__ q, const int8_t* __restrict__ w,
    const float* __restrict__ scales, const float* __restrict__ zeros,
    const float* __restrict__ sx, const float* __restrict__ zpv,
    const int* __restrict__ sumq, const int* __restrict__ rsw,
    float* __restrict__ out)
{
    const int lane = threadIdx.x & 31;
    const int wv   = threadIdx.x >> 5;
    const int mw   = wv & 3;
    const int nw   = wv >> 2;
    const int m0   = blockIdx.y * 128 + mw * 32;
    const int n0   = blockIdx.x * 128 + nw * 64;

    const int hi = lane >> 4;
    const int lm = lane & 15;

    v8i acc[2][4];
    const v8i vzero = {0, 0, 0, 0, 0, 0, 0, 0};
#pragma unroll
    for (int mt = 0; mt < 2; ++mt)
#pragma unroll
        for (int nt = 0; nt < 4; ++nt) acc[mt][nt] = vzero;

    const int8_t* aBase[2];
#pragma unroll
    for (int mt = 0; mt < 2; ++mt)
        aBase[mt] = q + (size_t)(m0 + mt * 16 + lm) * IN_F + (hi ? 16 : 0);
    const int8_t* bBase[4];
#pragma unroll
    for (int nt = 0; nt < 4; ++nt)
        bBase[nt] = w + (size_t)(n0 + nt * 16 + lm) * IN_F + (hi ? 16 : 0);

    for (int k0 = 0; k0 < IN_F; k0 += 64) {
        v8i a[2], b[4];
#pragma unroll
        for (int mt = 0; mt < 2; ++mt) {
            const int8_t* p = aBase[mt] + k0;
            v2i a0 = *(const v2i*)(p);
            v2i a1 = *(const v2i*)(p + 8);
            v2i a2 = *(const v2i*)(p + 32);
            v2i a3 = *(const v2i*)(p + 40);
            v8i av = {a0.x, a0.y, a1.x, a1.y, a2.x, a2.y, a3.x, a3.y};
            a[mt] = av;
        }
#pragma unroll
        for (int nt = 0; nt < 4; ++nt) {
            const int8_t* p = bBase[nt] + k0;
            v4i b0 = *(const v4i*)(p);
            v4i b1 = *(const v4i*)(p + 32);
            v8i bv = {b0.x, b0.y, b0.z, b0.w, b1.x, b1.y, b1.z, b1.w};
            b[nt] = bv;
        }
#pragma unroll
        for (int mt = 0; mt < 2; ++mt)
#pragma unroll
            for (int nt = 0; nt < 4; ++nt)
                acc[mt][nt] = __builtin_amdgcn_wmma_i32_16x16x64_iu8(
                     true, a[mt],  true, b[nt],
                    acc[mt][nt],  false,  false);
    }

#pragma unroll
    for (int nt = 0; nt < 4; ++nt) {
        const int   n  = n0 + nt * 16 + lm;
        const float sc = scales[n];
        const float zr = zeros[n];
        const float rw = (float)rsw[n];
#pragma unroll
        for (int mt = 0; mt < 2; ++mt) {
#pragma unroll
            for (int e = 0; e < 8; ++e) {
                const int   m    = m0 + mt * 16 + hi * 8 + e;
                const float zpf  = zpv[m];
                const float corr = (float)acc[mt][nt][e]
                                   - zpf * rw
                                   - zr * ((float)sumq[m] - (float)IN_F * zpf);
                out[(size_t)m * OUT_F + n] = sx[m] * sc * corr;
            }
        }
    }
}

extern "C" void kernel_launch(void* const* d_in, const int* in_sizes, int n_in,
                              void* d_out, int out_size, void* d_ws, size_t ws_size,
                              hipStream_t stream)
{
    const float*  x      = (const float*)d_in[0];
    const int8_t* w8     = (const int8_t*)d_in[1];
    const float*  scales = (const float*)d_in[2];
    const float*  zeros  = (const float*)d_in[3];
    float*        out    = (float*)d_out;

    uint8_t* ws = (uint8_t*)d_ws;
    size_t off = 0;
    int8_t* q    = (int8_t*)(ws + off); off += (size_t)TOKENS * IN_F;
    float*  sx   = (float*)(ws + off);  off += (size_t)TOKENS * 4;
    float*  zp   = (float*)(ws + off);  off += (size_t)TOKENS * 4;
    int*    sumq = (int*)(ws + off);    off += (size_t)TOKENS * 4;
    int*    rsw  = (int*)(ws + off);    off += (size_t)OUT_F * 4;

    quant_kernel<<<TOKENS, 256, 0, stream>>>(x, q, sx, zp, sumq);
    rowsum_kernel<<<OUT_F, 256, 0, stream>>>(w8, rsw);

    dim3 grid(OUT_F / 128, TOKENS / 128);
    gemm_kernel<<<grid, 256, 0, stream>>>(q, w8, scales, zeros, sx, zp, sumq, rsw, out);
}
